// VectorizedIntelligentNetwork_55585466745308
// MI455X (gfx1250) — compile-verified, matrix operand order checked by reading, not run on hardware
//
#include <hip/hip_runtime.h>

// ---------------------------------------------------------------------------
// Problem constants (from the reference)
// ---------------------------------------------------------------------------
#define B_    8
#define T_    64
#define IN_   128
#define OUT_  256
#define N_IN  32
#define N_TR  96
#define N_DYN 32
#define N_ACT 32
#define N_STAT 160      // N_IN + N_TR + N_ACT
#define NN    192       // N_STAT + N_DYN
#define IDX0  128       // N_IN + N_TR
#define H_    256
#define KQ_   256
#define S_    256

typedef __bf16 bf16;
typedef __bf16 v16bf __attribute__((ext_vector_type(16)));
typedef __bf16 v8bf  __attribute__((ext_vector_type(8)));
typedef float  v8f   __attribute__((ext_vector_type(8)));

#define PITCH 80   // LDS row pitch in elements; 160B rows, 16B-aligned for b128

// ---------------------------------------------------------------------------
// Generic batched GEMM with CDNA5 WMMA (bf16 in, fp32 accumulate)
//   C[z](MxN) = act( A[z](MxK) * B[z] + bias[z] (+C[z] if accum) )
// A: row-major MxK (k contiguous).  B: **NxK layout** (k contiguous), i.e.
// Bt[n*ldb + k].  All K multiples of 64.  Outputs: fp32 Cf, bf16 Cb (normal
// layout), bf16 Ct (arbitrary 3-stride transposed layout) — each optional.
// Block = 256 threads = 8 waves; 16x128 output macro-tile, one 16x16 WMMA
// tile per wave.  K is consumed 64/stage: A(16x64)+B(128x64) double-buffered
// in LDS, staged with global_load_b128 -> regs -> ds_store_b128; two WMMAs
// per barrier pair; fragments are two contiguous v8bf LDS reads per operand
// (ISA 16-bit 16x32 fragment layout: lanes 0-15 K{0..7,16..23},
//  lanes 16-31 K{8..15,24..31}).
// ---------------------------------------------------------------------------
__global__ __launch_bounds__(256) void gemm_wmma_bf16(
    const bf16* __restrict__ A,  long long sA, int lda,
    const bf16* __restrict__ Bt, long long sB, int ldb,
    const float* __restrict__ bias, long long sBias,
    float* __restrict__ Cf, bf16* __restrict__ Cb, long long sC, int ldc,
    bf16* __restrict__ Ct, long long tZ, long long tRow, long long tCol,
    int M, int N, int K, int act, int accum)
{
    __shared__ bf16 ldsA[2][16 * PITCH];
    __shared__ bf16 ldsB[2][128 * PITCH];

    const int z     = blockIdx.z;
    const int tileN = blockIdx.x * 128;
    const int tileM = blockIdx.y * 16;
    const int tid   = threadIdx.x;
    const int lane  = tid & 31;
    const int wave  = tid >> 5;
    const int half  = lane >> 4;
    const int mrem  = M - tileM;

    const bf16* Az = A  + (size_t)z * sA + (size_t)tileM * lda;
    const bf16* Bz = Bt + (size_t)z * sB;

    // staging assignments (each uint4 = 8 bf16; K-tile = 64)
    const int  arow = tid >> 3;                 // A: 16x64 = 128 uint4 (tid<128)
    const int  akc  = (tid & 7) * 8;
    const bool aval = (tid < 128) && (arow < mrem);
    const int  bcol = tid >> 3;                 // B: 128x64 = 1024 uint4, 4/thread
    const int  bkc  = (tid & 7) * 8;            //    cols bcol, bcol+32, +64, +96

    uint4 ra, rb[4];
    const uint4 zz = make_uint4(0u, 0u, 0u, 0u);

    auto gload = [&](int k0) {
        ra = zz;
        if (aval) ra = *(const uint4*)(Az + (size_t)arow * lda + k0 + akc);
        #pragma unroll
        for (int i = 0; i < 4; ++i) {
            rb[i] = zz;
            const int cg = tileN + bcol + i * 32;
            if (cg < N) rb[i] = *(const uint4*)(Bz + (size_t)cg * ldb + k0 + bkc);
        }
    };
    auto sstore = [&](int buf) {
        if (tid < 128) *(uint4*)&ldsA[buf][arow * PITCH + akc] = ra;
        #pragma unroll
        for (int i = 0; i < 4; ++i)
            *(uint4*)&ldsB[buf][(bcol + i * 32) * PITCH + bkc] = rb[i];
    };

    const int m      = lane & 15;
    const int colsel = wave * 16 + m;
    v8f acc = {};
    const int nk = K >> 6;

    gload(0);
    for (int it = 0; it < nk; ++it) {
        const int buf = it & 1;
        sstore(buf);
        __syncthreads();
        if (it + 1 < nk) gload((it + 1) << 6);   // overlap next stage's HBM reads

        #pragma unroll
        for (int kk = 0; kk < 64; kk += 32) {
            const bf16* pa = &ldsA[buf][m * PITCH + kk + half * 8];
            const bf16* pb = &ldsB[buf][colsel * PITCH + kk + half * 8];
            v8bf alo = *(const v8bf*)pa;
            v8bf ahi = *(const v8bf*)(pa + 16);
            v8bf blo = *(const v8bf*)pb;
            v8bf bhi = *(const v8bf*)(pb + 16);
            v16bf af = __builtin_shufflevector(alo, ahi, 0,1,2,3,4,5,6,7,8,9,10,11,12,13,14,15);
            v16bf bf = __builtin_shufflevector(blo, bhi, 0,1,2,3,4,5,6,7,8,9,10,11,12,13,14,15);
            acc = __builtin_amdgcn_wmma_f32_16x16x32_bf16(
                      false, af, false, bf, (short)0, acc, false, false);
        }
        __syncthreads();
    }

    // epilogue: D layout — VGPR r: M=r (lanes 0-15) / M=r+8 (lanes 16-31)
    const int colg = tileN + wave * 16 + m;
    if (colg < N) {
        float bv = bias ? bias[(size_t)z * sBias + colg] : 0.f;
        #pragma unroll
        for (int r = 0; r < 8; ++r) {
            int row = r + half * 8;
            if (row < mrem) {
                size_t off = (size_t)z * sC + (size_t)(tileM + row) * ldc + colg;
                float v = acc[r] + bv;
                if (accum) v += Cf[off];
                if (act == 1) v = fmaxf(v, 0.f);
                if (Cf) Cf[off] = v;
                if (Cb) Cb[off] = (bf16)v;
                if (Ct) Ct[(size_t)z * tZ + (size_t)(tileM + row) * tRow
                           + (size_t)colg * tCol] = (bf16)v;
            }
        }
    }
}

// ---------------------------------------------------------------------------
// Row softmax (with 1/sqrt(KQ) scale) -> bf16 output
// ---------------------------------------------------------------------------
__global__ __launch_bounds__(256) void softmax_rows(
    const float* __restrict__ in, bf16* __restrict__ out, int rowlen, float scale)
{
    const int row = blockIdx.x;
    const float* r = in + (size_t)row * rowlen;
    __shared__ float red[256];
    const int t = threadIdx.x;

    float m = -3.4e38f;
    for (int i = t; i < rowlen; i += 256) m = fmaxf(m, r[i] * scale);
    red[t] = m; __syncthreads();
    for (int s = 128; s > 0; s >>= 1) { if (t < s) red[t] = fmaxf(red[t], red[t + s]); __syncthreads(); }
    const float mx = red[0]; __syncthreads();

    float sum = 0.f;
    for (int i = t; i < rowlen; i += 256) sum += __expf(r[i] * scale - mx);
    red[t] = sum; __syncthreads();
    for (int s = 128; s > 0; s >>= 1) { if (t < s) red[t] += red[t + s]; __syncthreads(); }
    const float inv = 1.f / red[0];

    for (int i = t; i < rowlen; i += 256)
        out[(size_t)row * rowlen + i] = (bf16)(__expf(r[i] * scale - mx) * inv);
}

// ---------------------------------------------------------------------------
// LSTM pointwise: gates(B,N,4H) fp32 + c -> c_new fp32, h_new bf16 (in place)
// ---------------------------------------------------------------------------
__global__ __launch_bounds__(256) void lstm_pointwise(
    const float* __restrict__ gates, float* __restrict__ c, bf16* __restrict__ h_bf)
{
    const int idx = blockIdx.x * 256 + threadIdx.x;
    if (idx >= B_ * NN * H_) return;
    const int hI = idx & (H_ - 1);
    const int n  = (idx / H_) % NN;
    const int b  = idx / (H_ * NN);
    const size_t g = ((size_t)b * NN + n) * (4 * H_) + hI;
    const float ig = gates[g], fg = gates[g + H_], gg = gates[g + 2 * H_], og = gates[g + 3 * H_];
    const float si = 1.f / (1.f + __expf(-ig));
    const float sf = 1.f / (1.f + __expf(-fg));
    const float so = 1.f / (1.f + __expf(-og));
    const float cn = sf * c[idx] + si * tanhf(gg);
    const float hn = so * tanhf(cn);
    c[idx]    = cn;
    h_bf[idx] = (bf16)hn;
}

// ---------------------------------------------------------------------------
// Utility kernels
// ---------------------------------------------------------------------------
__global__ void convert_f32_bf16(const float* __restrict__ in, bf16* __restrict__ out, long long n)
{
    long long i = (long long)blockIdx.x * blockDim.x + threadIdx.x;
    const long long stride = (long long)gridDim.x * blockDim.x;
    for (; i < n; i += stride) out[i] = (bf16)in[i];
}

// Batched convert + transpose: out[(z*Nc + n)*Kd + k] = in[(z*Kd + k)*Nc + n]
// (weights KxN row-major  ->  NxK bf16, k-contiguous for the GEMM B operand)
__global__ void transpose_f32_bf16(const float* __restrict__ in, bf16* __restrict__ out,
                                   int Kd, int Nc, long long total)
{
    long long i = (long long)blockIdx.x * blockDim.x + threadIdx.x;
    const long long stride = (long long)gridDim.x * blockDim.x;
    for (; i < total; i += stride) {
        long long k   = i % Kd;
        long long rem = i / Kd;
        long long n   = rem % Nc;
        long long zq  = rem / Nc;
        out[i] = (bf16)in[(zq * Kd + k) * Nc + n];
    }
}

__global__ void zero_bf16(bf16* p, long long n)
{
    long long i = (long long)blockIdx.x * blockDim.x + threadIdx.x;
    const long long stride = (long long)gridDim.x * blockDim.x;
    for (; i < n; i += stride) p[i] = (bf16)0.f;
}

__global__ void zero_f32(float* p, long long n)
{
    long long i = (long long)blockIdx.x * blockDim.x + threadIdx.x;
    const long long stride = (long long)gridDim.x * blockDim.x;
    for (; i < n; i += stride) p[i] = 0.f;
}

// Broadcast static q/k (N_STAT x KQ) to all batches of Q_bf/K_bf (B,N,KQ)
__global__ void fill_static_qk(const float* __restrict__ q, const float* __restrict__ k,
                               bf16* __restrict__ Qb, bf16* __restrict__ Kb)
{
    const int idx = blockIdx.x * 256 + threadIdx.x;
    if (idx >= N_STAT * KQ_) return;
    const bf16 qv = (bf16)q[idx], kv = (bf16)k[idx];
    const int n = idx / KQ_, kk = idx % KQ_;
    #pragma unroll
    for (int b = 0; b < B_; ++b) {
        const size_t o = ((size_t)b * NN + n) * KQ_ + kk;
        Qb[o] = qv; Kb[o] = kv;
    }
}

// ---------------------------------------------------------------------------
// Host launcher
// ---------------------------------------------------------------------------
extern "C" void kernel_launch(void* const* d_in, const int* in_sizes, int n_in,
                              void* d_out, int out_size, void* d_ws, size_t ws_size,
                              hipStream_t stream)
{
    (void)in_sizes; (void)n_in; (void)out_size; (void)ws_size;
    const float* x         = (const float*)d_in[0];
    const float* w_rec_in  = (const float*)d_in[1];
    const float* b_rec_in  = (const float*)d_in[2];
    const float* w_rec_oth = (const float*)d_in[3];
    const float* b_rec_oth = (const float*)d_in[4];
    const float* w_ih      = (const float*)d_in[5];
    const float* b_ih      = (const float*)d_in[6];
    const float* w_hh      = (const float*)d_in[7];
    const float* b_hh      = (const float*)d_in[8];
    const float* q_static  = (const float*)d_in[9];
    const float* k_static  = (const float*)d_in[10];
    const float* w_qd      = (const float*)d_in[11];
    const float* b_qd      = (const float*)d_in[12];
    const float* w_kd      = (const float*)d_in[13];
    const float* b_kd      = (const float*)d_in[14];
    const float* w_snd     = (const float*)d_in[15];
    const float* b_snd     = (const float*)d_in[16];
    const float* w_proj    = (const float*)d_in[17];
    const float* b_proj    = (const float*)d_in[18];
    float* out = (float*)d_out;

    // ---- workspace carve-out (256B aligned) ----
    char* base = (char*)d_ws;
    size_t off = 0;
    auto take = [&](size_t bytes) { char* p = base + off; off += (bytes + 255) & ~(size_t)255; return p; };

    const long long nWri = (long long)N_IN * IN_ * H_;
    const long long nWro = (long long)(NN - N_IN) * S_ * H_;
    const long long nWih = (long long)NN * H_ * 4 * H_;
    const long long nWqd = (long long)N_DYN * H_ * KQ_;
    const long long nWsn = (long long)NN * H_ * S_;
    const long long nWpj = (long long)N_ACT * S_ * OUT_;
    const long long nX   = (long long)B_ * T_ * IN_;

    // all weight buffers hold the *transposed* (NxK, k-contiguous) bf16 form
    bf16* wriT = (bf16*)take(nWri * 2);
    bf16* wroT = (bf16*)take(nWro * 2);
    bf16* wihT = (bf16*)take(nWih * 2);
    bf16* whhT = (bf16*)take(nWih * 2);
    bf16* wqdT = (bf16*)take(nWqd * 2);
    bf16* wkdT = (bf16*)take(nWqd * 2);
    bf16* wsnT = (bf16*)take(nWsn * 2);
    bf16* wpjT = (bf16*)take(nWpj * 2);
    bf16* x_bf = (bf16*)take(nX * 2);

    const long long nBNH = (long long)B_ * NN * H_;
    const long long nBNS = (long long)B_ * NN * S_;
    const long long nAtt = (long long)B_ * NN * NN;
    bf16* h_bf    = (bf16*)take(nBNH * 2);
    bf16* prev_bf = (bf16*)take(nBNS * 2);   // (B,N,S) for out-proj A operand
    bf16* prevT   = (bf16*)take(nBNS * 2);   // (B,S,N) for ctx B operand
    bf16* Q_bf    = (bf16*)take((long long)B_ * NN * KQ_ * 2);
    bf16* K_bf    = (bf16*)take((long long)B_ * NN * KQ_ * 2);
    bf16* attn_bf = (bf16*)take(nAtt * 2);
    bf16* ctx_bf  = (bf16*)take(nBNS * 2);
    bf16* lin_bf  = (bf16*)take(nBNH * 2);
    float* c_f    = (float*)take(nBNH * 4);
    float* scores = (float*)take(nAtt * 4);
    float* gates  = (float*)take((long long)B_ * NN * 4 * H_ * 4);

    // ---- one-time conversions (weights transposed to NxK) & init ----
    transpose_f32_bf16<<<4096, 256, 0, stream>>>(w_rec_in,  wriT, IN_, H_,     nWri);
    transpose_f32_bf16<<<4096, 256, 0, stream>>>(w_rec_oth, wroT, S_,  H_,     nWro);
    transpose_f32_bf16<<<8192, 256, 0, stream>>>(w_ih,      wihT, H_,  4 * H_, nWih);
    transpose_f32_bf16<<<8192, 256, 0, stream>>>(w_hh,      whhT, H_,  4 * H_, nWih);
    transpose_f32_bf16<<<4096, 256, 0, stream>>>(w_qd,      wqdT, H_,  KQ_,    nWqd);
    transpose_f32_bf16<<<4096, 256, 0, stream>>>(w_kd,      wkdT, H_,  KQ_,    nWqd);
    transpose_f32_bf16<<<4096, 256, 0, stream>>>(w_snd,     wsnT, H_,  S_,     nWsn);
    transpose_f32_bf16<<<4096, 256, 0, stream>>>(w_proj,    wpjT, N_ACT * S_, OUT_, nWpj);
    convert_f32_bf16  <<<256,  256, 0, stream>>>(x, x_bf, nX);
    zero_bf16<<<1536, 256, 0, stream>>>(h_bf,    nBNH);
    zero_bf16<<<1536, 256, 0, stream>>>(prev_bf, nBNS);
    zero_bf16<<<1536, 256, 0, stream>>>(prevT,   nBNS);
    zero_f32 <<<1536, 256, 0, stream>>>(c_f,     nBNH);
    fill_static_qk<<<(N_STAT * KQ_ + 255) / 256, 256, 0, stream>>>(q_static, k_static, Q_bf, K_bf);

    const float scale = 0.0625f;  // 1/sqrt(256)

    for (int t = 0; t < T_; ++t) {
        // 1/2) dynamic q,k : per n in N_DYN, (8xH)@(HxKQ)
        gemm_wmma_bf16<<<dim3(2, 1, N_DYN), 256, 0, stream>>>(
            h_bf + (size_t)N_STAT * H_, H_, NN * H_,
            wqdT, (long long)KQ_ * H_, H_,
            b_qd, KQ_,
            nullptr, Q_bf + (size_t)N_STAT * KQ_, KQ_, NN * KQ_,
            nullptr, 0, 0, 0,
            B_, KQ_, H_, 0, 0);
        gemm_wmma_bf16<<<dim3(2, 1, N_DYN), 256, 0, stream>>>(
            h_bf + (size_t)N_STAT * H_, H_, NN * H_,
            wkdT, (long long)KQ_ * H_, H_,
            b_kd, KQ_,
            nullptr, K_bf + (size_t)N_STAT * KQ_, KQ_, NN * KQ_,
            nullptr, 0, 0, 0,
            B_, KQ_, H_, 0, 0);
        // 3) scores = Q @ K^T  (K_bf is naturally NxK for this product)
        gemm_wmma_bf16<<<dim3(2, 12, B_), 256, 0, stream>>>(
            Q_bf, (long long)NN * KQ_, KQ_,
            K_bf, (long long)NN * KQ_, KQ_,
            nullptr, 0,
            scores, nullptr, (long long)NN * NN, NN,
            nullptr, 0, 0, 0,
            NN, NN, KQ_, 0, 0);
        // 4) softmax (scaled) -> attn bf16
        softmax_rows<<<B_ * NN, 256, 0, stream>>>(scores, attn_bf, NN, scale);
        // 5) ctx = attn @ prev   (B operand = prevT, (B,S,N) k-contiguous)
        gemm_wmma_bf16<<<dim3(2, 12, B_), 256, 0, stream>>>(
            attn_bf, (long long)NN * NN, NN,
            prevT, (long long)S_ * NN, NN,
            nullptr, 0,
            nullptr, ctx_bf, (long long)NN * S_, S_,
            nullptr, 0, 0, 0,
            NN, S_, NN, 0, 0);
        // 6) rec_in = relu(x_t @ w_rec_input + b), A shared across modules
        gemm_wmma_bf16<<<dim3(2, 1, N_IN), 256, 0, stream>>>(
            x_bf + (size_t)t * IN_, 0, T_ * IN_,
            wriT, (long long)H_ * IN_, IN_,
            b_rec_in, H_,
            nullptr, lin_bf, H_, NN * H_,
            nullptr, 0, 0, 0,
            B_, H_, IN_, 1, 0);
        // 7) rec_oth = relu(ctx[:,N_IN:,:] @ w_rec_other + b)
        gemm_wmma_bf16<<<dim3(2, 1, NN - N_IN), 256, 0, stream>>>(
            ctx_bf + (size_t)N_IN * S_, S_, NN * S_,
            wroT, (long long)H_ * S_, S_,
            b_rec_oth, H_,
            nullptr, lin_bf + (size_t)N_IN * H_, H_, NN * H_,
            nullptr, 0, 0, 0,
            B_, H_, S_, 1, 0);
        // 8) gates = lstm_in @ w_ih + b_ih
        gemm_wmma_bf16<<<dim3(8, 1, NN), 256, 0, stream>>>(
            lin_bf, H_, NN * H_,
            wihT, (long long)(4 * H_) * H_, H_,
            b_ih, 4 * H_,
            gates, nullptr, 4 * H_, NN * 4 * H_,
            nullptr, 0, 0, 0,
            B_, 4 * H_, H_, 0, 0);
        // 9) gates += h @ w_hh + b_hh
        gemm_wmma_bf16<<<dim3(8, 1, NN), 256, 0, stream>>>(
            h_bf, H_, NN * H_,
            whhT, (long long)(4 * H_) * H_, H_,
            b_hh, 4 * H_,
            gates, nullptr, 4 * H_, NN * 4 * H_,
            nullptr, 0, 0, 0,
            B_, 4 * H_, H_, 0, 1);
        // 10) pointwise LSTM update (h_bf becomes h_new)
        lstm_pointwise<<<(B_ * NN * H_ + 255) / 256, 256, 0, stream>>>(gates, c_f, h_bf);
        // 11) prev_new = h_new @ w_sender + b_sender
        //     writes prev_bf (B,N,S) AND prevT (B,S,N) via transposed path
        gemm_wmma_bf16<<<dim3(2, 1, NN), 256, 0, stream>>>(
            h_bf, H_, NN * H_,
            wsnT, (long long)S_ * H_, H_,
            b_snd, S_,
            nullptr, prev_bf, S_, NN * S_,
            prevT, 1, (long long)S_ * NN, NN,
            B_, S_, H_, 0, 0);
        // 12) out_t = prev_new[:,IDX0:IDX0+N_ACT,:].reshape(8,8192) @ w_proj + b_proj
        gemm_wmma_bf16<<<dim3(2, 1, 1), 256, 0, stream>>>(
            prev_bf + (size_t)IDX0 * S_, 0, NN * S_,
            wpjT, 0, N_ACT * S_,
            b_proj, 0,
            out + (size_t)t * OUT_, nullptr, 0, T_ * OUT_,
            nullptr, 0, 0, 0,
            B_, OUT_, N_ACT * S_, 0, 0);
    }
}
